// EncoderBlock_91250875171620
// MI455X (gfx1250) — matrix operand order repaired and checked by reading, not compiled, not run on hardware
//
#include <hip/hip_runtime.h>
#include <hip/hip_bf16.h>

// ---------------------------------------------------------------------------
// Transformer encoder block for MI455X (gfx1250, wave32, WMMA bf16).
// GEMMs: V_WMMA_F32_16X16X32_BF16 fed from LDS tiles staged with
// GLOBAL_LOAD_ASYNC_TO_LDS_B128 (ASYNCcnt), double buffered.
// ---------------------------------------------------------------------------

typedef __attribute__((ext_vector_type(16))) __bf16 v16bf;
typedef __attribute__((ext_vector_type(8)))  __bf16 v8bf;
typedef __attribute__((ext_vector_type(8)))  float  v8f;

union BF16x16 { v16bf v; v8bf h[2]; };

#define BB   2
#define TT   2048
#define CC   1024
#define HH   16
#define HSZ  64
#define MTOT (BB*TT)          // 4096 rows
#define FF   (4*CC)           // 4096

#define LROW 40               // padded LDS row (32 data + 8 pad) -> 80B stride
#define TILEELEMS (128*LROW)  // one staged tile (A or B), elements

static __device__ __forceinline__ v8f wmma_bf16(const v16bf& a, const v16bf& b, const v8f& c) {
    return __builtin_amdgcn_wmma_f32_16x16x32_bf16(false, a, false, b, (short)0, c, false, false);
}

// Async global->LDS 16B copy (per lane). VDST = LDS byte address (wave-relative,
// equals low 32 bits of the generic LDS pointer), ADDR = 64-bit global address.
static __device__ __forceinline__ void async_copy_b128(const __bf16* g, unsigned lds_byte) {
    asm volatile("global_load_async_to_lds_b128 %0, %1, off"
                 :: "v"(lds_byte), "v"(g) : "memory");
}
static __device__ __forceinline__ void wait_async() {
    asm volatile("s_wait_asynccnt 0" ::: "memory");
}

// ---------------------------------------------------------------------------
// Double-buffered LDS GEMM mainloop.
// A  : [M,K] bf16 row-major (activations)
// Bt : [N,K] bf16 row-major (pre-transposed weights; row n = output column n)
// Block tile 128x128 (8 waves, wave tile 32x64 = 2x4 WMMA accums), K-step 32.
// LDS tiles 128x32 (rows padded to 40 elems -> conflict-free ds_load_b128).
// Fragment layouts (wave32):
//   A 16x32 : lane l -> row (l&15), K = (l>>4)*8 + {0..7} and +16..+23
//   B 32x16 : lane l -> col (l&15), K = (l>>4)*8 + {0..7} and +16..+23
// ---------------------------------------------------------------------------
static __device__ __forceinline__ void gemm_lds(
    const __bf16* __restrict__ A, int lda,
    const __bf16* __restrict__ Bt, int ldb,
    int K, int blockM, int blockN,
    __bf16* smA, __bf16* smB,            // each [2][TILEELEMS]
    v8f (&acc)[2][4])
{
    const int tid  = threadIdx.x;
    const int lane = tid & 31, wave = tid >> 5;
    const int wm = wave & 3, wn = wave >> 2;
    const int half = lane >> 4, l16 = lane & 15;

    const unsigned aoff = (unsigned)(uintptr_t)smA;
    const unsigned boff = (unsigned)(uintptr_t)smB;
    // 512 16B-chunks per 8KB tile; thread handles chunks tid and tid+256.
    const int r0 = tid >> 2;            // row of first chunk (0..63)
    const int c0 = (tid & 3) * 8;       // element offset within row (0/8/16/24)

    auto stage = [&](int buf, int k0) {
        const __bf16* Ab = A  + (size_t)blockM * lda + k0;
        const __bf16* Bb = Bt + (size_t)blockN * ldb + k0;
        const unsigned la = aoff + (unsigned)buf * (TILEELEMS * 2);
        const unsigned lb = boff + (unsigned)buf * (TILEELEMS * 2);
        async_copy_b128(Ab + (size_t)r0        * lda + c0, la + r0 * 80 + c0 * 2);
        async_copy_b128(Ab + (size_t)(r0 + 64) * lda + c0, la + (r0 + 64) * 80 + c0 * 2);
        async_copy_b128(Bb + (size_t)r0        * ldb + c0, lb + r0 * 80 + c0 * 2);
        async_copy_b128(Bb + (size_t)(r0 + 64) * ldb + c0, lb + (r0 + 64) * 80 + c0 * 2);
    };

    stage(0, 0);
    wait_async();
    __syncthreads();

    int buf = 0;
    for (int k0 = 0; k0 < K; k0 += 32, buf ^= 1) {
        const bool more = (k0 + 32) < K;
        if (more) stage(buf ^ 1, k0 + 32);   // overlap next copies with compute

        const __bf16* sa = smA + buf * TILEELEMS;
        const __bf16* sb = smB + buf * TILEELEMS;
        BF16x16 a[2], b[4];
#pragma unroll
        for (int i = 0; i < 2; ++i) {
            const __bf16* p = sa + (wm*32 + i*16 + l16) * LROW + half*8;
            a[i].h[0] = *(const v8bf*)p;
            a[i].h[1] = *(const v8bf*)(p + 16);
        }
#pragma unroll
        for (int j = 0; j < 4; ++j) {
            const __bf16* p = sb + (wn*64 + j*16 + l16) * LROW + half*8;
            b[j].h[0] = *(const v8bf*)p;
            b[j].h[1] = *(const v8bf*)(p + 16);
        }
#pragma unroll
        for (int i = 0; i < 2; ++i)
#pragma unroll
            for (int j = 0; j < 4; ++j)
                acc[i][j] = wmma_bf16(a[i].v, b[j].v, acc[i][j]);

        if (more) wait_async();
        __syncthreads();
    }
}

// ---------------------------------------------------------------------------
// LayerNorm (row = C=1024 floats) -> bf16
// ---------------------------------------------------------------------------
__global__ __launch_bounds__(256) void ln_bf16_kernel(
    const float* __restrict__ x, const float* __restrict__ w,
    const float* __restrict__ b, __bf16* __restrict__ y)
{
    const int row = blockIdx.x;
    const int tid = threadIdx.x;
    const float* xr = x + (size_t)row * CC;
    float s = 0.f, sq = 0.f;
    for (int i = tid; i < CC; i += 256) { float v = xr[i]; s += v; sq += v*v; }
    __shared__ float rs[256], rq[256];
    rs[tid] = s; rq[tid] = sq; __syncthreads();
    for (int off = 128; off > 0; off >>= 1) {
        if (tid < off) { rs[tid] += rs[tid+off]; rq[tid] += rq[tid+off]; }
        __syncthreads();
    }
    const float mu  = rs[0] * (1.f/CC);
    const float var = rq[0] * (1.f/CC) - mu*mu;
    const float inv = rsqrtf(var + 1e-5f);
    __bf16* yr = y + (size_t)row * CC;
    for (int i = tid; i < CC; i += 256)
        yr[i] = (__bf16)((xr[i] - mu) * inv * w[i] + b[i]);
}

// ---------------------------------------------------------------------------
// Weight prep: dst[n*K + k] = (bf16) src[k*N + n]   (Bt layout), K power of 2
// ---------------------------------------------------------------------------
__global__ void transpose_bf16_kernel(const float* __restrict__ src,
                                      __bf16* __restrict__ dst,
                                      int N, int K, int kshift)
{
    size_t tot = (size_t)N * K;
    for (size_t idx = (size_t)blockIdx.x * blockDim.x + threadIdx.x;
         idx < tot; idx += (size_t)gridDim.x * blockDim.x) {
        int k = (int)(idx & (size_t)(K - 1));
        int n = (int)(idx >> kshift);
        dst[idx] = (__bf16)src[(size_t)k * N + n];
    }
}

// Wq/Wk/Wv [H,C,HS] -> stacked Bt [3C, C]: row (sel*1024 + h*64 + d), col c
__global__ void qkv_weight_kernel(const float* __restrict__ Wq,
                                  const float* __restrict__ Wk,
                                  const float* __restrict__ Wv,
                                  __bf16* __restrict__ dst)
{
    size_t tot = (size_t)3 * CC * CC;
    for (size_t idx = (size_t)blockIdx.x * blockDim.x + threadIdx.x;
         idx < tot; idx += (size_t)gridDim.x * blockDim.x) {
        int c   = (int)(idx & (CC - 1));
        int n   = (int)(idx >> 10);
        int sel = n >> 10;
        int nn  = n & (CC - 1);
        int h   = nn >> 6;
        int d   = nn & (HSZ - 1);
        const float* W = (sel == 0) ? Wq : (sel == 1) ? Wk : Wv;
        dst[idx] = (__bf16)W[((size_t)h * CC + c) * HSZ + d];
    }
}

// ---------------------------------------------------------------------------
// Fused QKV GEMM: xn[4096,1024] @ Wqkv_t -> q,k [B,H,T,HS] bf16, v^T [B,H,HS,T]
// ---------------------------------------------------------------------------
__global__ __launch_bounds__(256) void qkv_gemm_kernel(
    const __bf16* __restrict__ xn, const __bf16* __restrict__ wt,
    const float* __restrict__ bq, const float* __restrict__ bk,
    const float* __restrict__ bv,
    __bf16* __restrict__ q, __bf16* __restrict__ k, __bf16* __restrict__ vt)
{
    __shared__ __align__(16) __bf16 smA[2 * TILEELEMS];
    __shared__ __align__(16) __bf16 smB[2 * TILEELEMS];
    const int lane = threadIdx.x & 31, wave = threadIdx.x >> 5;
    const int wm = wave & 3, wn = wave >> 2;
    const int m0 = blockIdx.x * 128 + wm * 32;
    const int n0 = blockIdx.y * 128 + wn * 64;
    const int half = lane >> 4, l16 = lane & 15;
    v8f acc[2][4] = {};
    gemm_lds(xn, CC, wt, CC, CC, blockIdx.x * 128, blockIdx.y * 128, smA, smB, acc);

#pragma unroll
    for (int j = 0; j < 4; ++j) {
        const int n   = n0 + j*16 + l16;
        const int sel = n >> 10;
        const int nn  = n & (CC - 1);
        const int h   = nn >> 6;
        const int d   = nn & (HSZ - 1);
        const float* bp = (sel == 0) ? bq : (sel == 1) ? bk : bv;
        const float bias = bp[nn];
#pragma unroll
        for (int i = 0; i < 2; ++i) {
            const int mbase = m0 + i*16 + half*8;
            const int bidx  = mbase >> 11;        // / T
            const int t     = mbase & (TT - 1);
            if (sel < 2) {
                __bf16* dstp = ((sel == 0) ? q : k) +
                               (((size_t)(bidx*HH + h) * TT + t) * HSZ + d);
#pragma unroll
                for (int r = 0; r < 8; ++r)
                    dstp[(size_t)r * HSZ] = (__bf16)(acc[i][j][r] + bias);
            } else {
                __bf16* dstp = vt + ((size_t)(bidx*HH + h) * HSZ + d) * TT + t;
                v8bf pk;
#pragma unroll
                for (int r = 0; r < 8; ++r) pk[r] = (__bf16)(acc[i][j][r] + bias);
                *(v8bf*)dstp = pk;               // rows are consecutive t
            }
        }
    }
}

// ---------------------------------------------------------------------------
// Flash attention. block = 4 waves; wave owns 16 Q rows; KV chunks of 32.
// S = Q.K^T via WMMA, online softmax, P transposed via per-wave LDS bounce,
// O += P.V via WMMA (V pre-transposed).  scale = C^-0.5 = 1/32.
// ---------------------------------------------------------------------------
__global__ __launch_bounds__(128) void attn_kernel(
    const __bf16* __restrict__ q, const __bf16* __restrict__ k,
    const __bf16* __restrict__ vt, __bf16* __restrict__ o)
{
    __shared__ __bf16 smem[4][16*32];
    const int lane = threadIdx.x & 31, wave = threadIdx.x >> 5;
    const int half = lane >> 4, l16 = lane & 15;
    const int bh = blockIdx.y;
    const int b  = bh >> 4, h = bh & (HH - 1);
    const int t0 = blockIdx.x * 64 + wave * 16;

    const __bf16* qb = q  + (size_t)bh * TT * HSZ;
    const __bf16* kb = k  + (size_t)bh * TT * HSZ;
    const __bf16* vb = vt + (size_t)bh * HSZ * TT;

    BF16x16 aq[2];
#pragma unroll
    for (int f = 0; f < 2; ++f) {
        const __bf16* p = qb + (size_t)(t0 + l16) * HSZ + f*32 + half*8;
        aq[f].h[0] = *(const v8bf*)p;
        aq[f].h[1] = *(const v8bf*)(p + 16);
    }

    v8f oacc[4] = {};
    float mrow[8], lrow[8];
#pragma unroll
    for (int r = 0; r < 8; ++r) { mrow[r] = -INFINITY; lrow[r] = 0.f; }
    const float scale = 0.03125f;   // 1/sqrt(C)

    for (int j = 0; j < TT; j += 32) {
        v8f s[2] = {};
#pragma unroll
        for (int nt = 0; nt < 2; ++nt)
#pragma unroll
            for (int f = 0; f < 2; ++f) {
                BF16x16 bk;
                const __bf16* p = kb + (size_t)(j + nt*16 + l16) * HSZ + f*32 + half*8;
                bk.h[0] = *(const v8bf*)p;
                bk.h[1] = *(const v8bf*)(p + 16);
                s[nt] = wmma_bf16(aq[f].v, bk.v, s[nt]);
            }

        float alpha[8];
#pragma unroll
        for (int r = 0; r < 8; ++r) {
            float s0 = s[0][r] * scale, s1 = s[1][r] * scale;
            float mx = fmaxf(s0, s1);
#pragma unroll
            for (int off = 8; off >= 1; off >>= 1)
                mx = fmaxf(mx, __shfl_xor(mx, off, 16));
            const float mnew = fmaxf(mrow[r], mx);
            alpha[r] = __expf(mrow[r] - mnew);
            mrow[r]  = mnew;
            const float p0 = __expf(s0 - mnew);
            const float p1 = __expf(s1 - mnew);
            s[0][r] = p0; s[1][r] = p1;
            float rsum = p0 + p1;
#pragma unroll
            for (int off = 8; off >= 1; off >>= 1)
                rsum += __shfl_xor(rsum, off, 16);
            lrow[r] = lrow[r] * alpha[r] + rsum;
#pragma unroll
            for (int nt = 0; nt < 4; ++nt) oacc[nt][r] *= alpha[r];
        }

        // transpose P (C layout -> A layout) through per-wave LDS bounce
        __bf16* ps = &smem[wave][0];
#pragma unroll
        for (int r = 0; r < 8; ++r) {
            const int row = half*8 + r;
            ps[row*32 + l16]      = (__bf16)s[0][r];
            ps[row*32 + 16 + l16] = (__bf16)s[1][r];
        }
        BF16x16 pa;   // same-wave DS ops are in-order
        const __bf16* pr = ps + l16*32 + half*8;
        pa.h[0] = *(const v8bf*)pr;
        pa.h[1] = *(const v8bf*)(pr + 16);

#pragma unroll
        for (int nt = 0; nt < 4; ++nt) {
            BF16x16 bv_;
            const __bf16* p = vb + (size_t)(nt*16 + l16) * TT + j + half*8;
            bv_.h[0] = *(const v8bf*)p;
            bv_.h[1] = *(const v8bf*)(p + 16);
            oacc[nt] = wmma_bf16(pa.v, bv_.v, oacc[nt]);
        }
    }

    float inv[8];
#pragma unroll
    for (int r = 0; r < 8; ++r) inv[r] = 1.f / lrow[r];
#pragma unroll
    for (int nt = 0; nt < 4; ++nt) {
        const int col = h*HSZ + nt*16 + l16;
#pragma unroll
        for (int r = 0; r < 8; ++r) {
            const int m = t0 + half*8 + r;
            o[(size_t)(b*TT + m) * CC + col] = (__bf16)(oacc[nt][r] * inv[r]);
        }
    }
}

// ---------------------------------------------------------------------------
// GEMM + bias + residual -> f32   (attention proj, FFN2)
// ---------------------------------------------------------------------------
__global__ __launch_bounds__(256) void gemm_bias_res_kernel(
    const __bf16* __restrict__ A, const __bf16* __restrict__ Bt,
    const float* __restrict__ bias, const float* __restrict__ resid,
    float* __restrict__ out, int N, int K, int lda, int ldb)
{
    __shared__ __align__(16) __bf16 smA[2 * TILEELEMS];
    __shared__ __align__(16) __bf16 smB[2 * TILEELEMS];
    const int lane = threadIdx.x & 31, wave = threadIdx.x >> 5;
    const int wm = wave & 3, wn = wave >> 2;
    const int m0 = blockIdx.x * 128 + wm * 32;
    const int n0 = blockIdx.y * 128 + wn * 64;
    const int half = lane >> 4, l16 = lane & 15;
    v8f acc[2][4] = {};
    gemm_lds(A, lda, Bt, ldb, K, blockIdx.x * 128, blockIdx.y * 128, smA, smB, acc);
#pragma unroll
    for (int j = 0; j < 4; ++j) {
        const int n = n0 + j*16 + l16;
        const float bv = bias[n];
#pragma unroll
        for (int i = 0; i < 2; ++i) {
#pragma unroll
            for (int r = 0; r < 8; ++r) {
                const int m = m0 + i*16 + half*8 + r;
                out[(size_t)m * N + n] = resid[(size_t)m * N + n] + acc[i][j][r] + bv;
            }
        }
    }
}

// ---------------------------------------------------------------------------
// GEMM + bias + exact-erf GELU -> bf16   (FFN1)
// ---------------------------------------------------------------------------
__global__ __launch_bounds__(256) void gemm_bias_gelu_kernel(
    const __bf16* __restrict__ A, const __bf16* __restrict__ Bt,
    const float* __restrict__ bias, __bf16* __restrict__ out,
    int N, int K, int lda, int ldb)
{
    __shared__ __align__(16) __bf16 smA[2 * TILEELEMS];
    __shared__ __align__(16) __bf16 smB[2 * TILEELEMS];
    const int lane = threadIdx.x & 31, wave = threadIdx.x >> 5;
    const int wm = wave & 3, wn = wave >> 2;
    const int m0 = blockIdx.x * 128 + wm * 32;
    const int n0 = blockIdx.y * 128 + wn * 64;
    const int half = lane >> 4, l16 = lane & 15;
    v8f acc[2][4] = {};
    gemm_lds(A, lda, Bt, ldb, K, blockIdx.x * 128, blockIdx.y * 128, smA, smB, acc);
#pragma unroll
    for (int j = 0; j < 4; ++j) {
        const int n = n0 + j*16 + l16;
        const float bv = bias[n];
#pragma unroll
        for (int i = 0; i < 2; ++i) {
#pragma unroll
            for (int r = 0; r < 8; ++r) {
                const int m = m0 + i*16 + half*8 + r;
                const float x = acc[i][j][r] + bv;
                const float g = 0.5f * x * (1.f + erff(x * 0.70710678118654752f));
                out[(size_t)m * N + n] = (__bf16)g;
            }
        }
    }
}

// ---------------------------------------------------------------------------
// Launch
// ---------------------------------------------------------------------------
extern "C" void kernel_launch(void* const* d_in, const int* in_sizes, int n_in,
                              void* d_out, int out_size, void* d_ws, size_t ws_size,
                              hipStream_t stream)
{
    const float* x    = (const float*)d_in[0];
    const float* Wq   = (const float*)d_in[1];
    const float* bq   = (const float*)d_in[2];
    const float* Wk   = (const float*)d_in[3];
    const float* bk   = (const float*)d_in[4];
    const float* Wv   = (const float*)d_in[5];
    const float* bv   = (const float*)d_in[6];
    const float* Wp   = (const float*)d_in[7];
    const float* bp   = (const float*)d_in[8];
    const float* W1   = (const float*)d_in[9];
    const float* b1   = (const float*)d_in[10];
    const float* W2   = (const float*)d_in[11];
    const float* b2   = (const float*)d_in[12];
    const float* ln1w = (const float*)d_in[13];
    const float* ln1b = (const float*)d_in[14];
    const float* ln2w = (const float*)d_in[15];
    const float* ln2b = (const float*)d_in[16];
    float* outp = (float*)d_out;

    char* wsb = (char*)d_ws;
    size_t off = 0;
    auto alloc = [&](size_t bytes) -> void* {
        void* p = wsb + off;
        off += (bytes + 255) & ~(size_t)255;
        return p;
    };
    __bf16* xn    = (__bf16*)alloc((size_t)MTOT * CC * 2);
    __bf16* onrm  = (__bf16*)alloc((size_t)MTOT * CC * 2);
    __bf16* wqkvT = (__bf16*)alloc((size_t)3 * CC * CC * 2);
    __bf16* wpT   = (__bf16*)alloc((size_t)CC * CC * 2);
    __bf16* w1T   = (__bf16*)alloc((size_t)FF * CC * 2);
    __bf16* w2T   = (__bf16*)alloc((size_t)CC * FF * 2);
    __bf16* qbuf  = (__bf16*)alloc((size_t)MTOT * CC * 2);
    __bf16* kbuf  = (__bf16*)alloc((size_t)MTOT * CC * 2);
    __bf16* vtbuf = (__bf16*)alloc((size_t)MTOT * CC * 2);
    __bf16* obuf  = (__bf16*)alloc((size_t)MTOT * CC * 2);
    float*  out1  = (float*) alloc((size_t)MTOT * CC * 4);
    __bf16* hbuf  = (__bf16*)alloc((size_t)MTOT * FF * 2);

    // 1) weight prep (bf16 + Bt layout)
    qkv_weight_kernel<<<2048, 256, 0, stream>>>(Wq, Wk, Wv, wqkvT);
    transpose_bf16_kernel<<<2048, 256, 0, stream>>>(Wp, wpT, CC, CC, 10);
    transpose_bf16_kernel<<<2048, 256, 0, stream>>>(W1, w1T, FF, CC, 10);
    transpose_bf16_kernel<<<2048, 256, 0, stream>>>(W2, w2T, CC, FF, 12);

    // 2) LN1
    ln_bf16_kernel<<<MTOT, 256, 0, stream>>>(x, ln1w, ln1b, xn);

    // 3) fused QKV projection (M=4096, N=3072, K=1024)
    qkv_gemm_kernel<<<dim3(MTOT/128, (3*CC)/128), 256, 0, stream>>>(
        xn, wqkvT, bq, bk, bv, qbuf, kbuf, vtbuf);

    // 4) attention (flash, online softmax)
    attn_kernel<<<dim3(TT/64, BB*HH), 128, 0, stream>>>(qbuf, kbuf, vtbuf, obuf);

    // 5) output projection + residual  (out1 = x + O@Wp + bp)
    gemm_bias_res_kernel<<<dim3(MTOT/128, CC/128), 256, 0, stream>>>(
        obuf, wpT, bp, x, out1, CC, CC, CC, CC);

    // 6) LN2
    ln_bf16_kernel<<<MTOT, 256, 0, stream>>>(out1, ln2w, ln2b, onrm);

    // 7) FFN1 + GELU (M=4096, N=4096, K=1024)
    gemm_bias_gelu_kernel<<<dim3(MTOT/128, FF/128), 256, 0, stream>>>(
        onrm, w1T, b1, hbuf, FF, CC, CC, CC);

    // 8) FFN2 + residual -> d_out (M=4096, N=1024, K=4096)
    gemm_bias_res_kernel<<<dim3(MTOT/128, CC/128), 256, 0, stream>>>(
        hbuf, w2T, b2, out1, outp, CC, FF, FF, FF);
}
